// HtrgGraphAttentionLayer_80152679678050
// MI455X (gfx1250) — matrix operand order repaired and checked by reading, not compiled, not run on hardware
//
#include <hip/hip_runtime.h>
#include <math.h>

typedef __attribute__((ext_vector_type(16))) __bf16 v16bf;
typedef __attribute__((ext_vector_type(8)))  float  v8f;

#define NTOT   256      // N1+N2
#define DK     64       // DIN == DOUT
#define BATCH  16
#define ROWS   (BATCH*NTOT)          // 4096
#define HELEMS (ROWS*DK)             // 262144
#define XSTR   68                    // LDS row stride (bank-conflict pad)

__device__ __forceinline__ float fast_tanh(float x) {
#if defined(__HIP_DEVICE_COMPILE__)
# if __has_builtin(__builtin_amdgcn_tanhf)
    return __builtin_amdgcn_tanhf(x);          // v_tanh_f32 (gfx1250 TRANS op)
# else
    float r; asm("v_tanh_f32 %0, %1" : "=v"(r) : "v"(x)); return r;
# endif
#else
    return tanhf(x);
#endif
}

// ---------------------------------------------------------------- projections
__global__ void proj_kernel(const float* __restrict__ x1, const float* __restrict__ x2,
                            const float* __restrict__ Wt1, const float* __restrict__ bt1,
                            const float* __restrict__ Wt2, const float* __restrict__ bt2,
                            float* __restrict__ x_f, __bf16* __restrict__ xh) {
    int idx = blockIdx.x * blockDim.x + threadIdx.x;
    if (idx >= HELEMS) return;
    int o = idx & 63, row = idx >> 6, n = row & 255, b = row >> 8;
    const float *in, *W, *bi;
    if (n < 128) { in = x1 + (b*128 + n)*DK;        W = Wt1; bi = bt1; }
    else         { in = x2 + (b*128 + (n-128))*DK;  W = Wt2; bi = bt2; }
    const float4* in4 = (const float4*)in;
    const float4* W4  = (const float4*)(W + o*DK);
    float acc = bi[o];
#pragma unroll 4
    for (int q = 0; q < DK/4; ++q) {
        const float4 a = in4[q], w = W4[q];
        acc += a.x*w.x + a.y*w.y + a.z*w.z + a.w*w.w;
    }
    x_f[idx] = acc;
    xh[idx]  = (__bf16)acc;
}

// ---------------------------------------------------------------- master mean
__global__ void mean_kernel(const float* __restrict__ x_f, float* __restrict__ master) {
    int b = blockIdx.x, k = threadIdx.x;   // 64 threads
    float s = 0.f;
    for (int n = 0; n < NTOT; ++n) s += x_f[(b*NTOT + n)*DK + k];
    master[b*DK + k] = s * (1.f/256.f);
}

// ------------------------------------------------- pairwise attention (WMMA)
// block = 256 thr (8 waves); grid = (16 batches, 32 i-octets); wave handles one i.
__global__ __launch_bounds__(256) void attn_kernel(
        const __bf16* __restrict__ xh,
        const float* __restrict__ Watt, const float* __restrict__ batt,
        const float* __restrict__ w11,  const float* __restrict__ w22,
        const float* __restrict__ w12,
        const float* __restrict__ Wpa,  const float* __restrict__ bpa,
        const float* __restrict__ Wpo,  const float* __restrict__ bpo,
        float* __restrict__ h_ws) {
    __shared__ __bf16 Xs[NTOT][XSTR];     // X_b tile (bf16)
    __shared__ __bf16 Ws[DK][XSTR];       // Watt (bf16)
    __shared__ float  sBuf[8][NTOT];      // per-wave scores / amap
    __shared__ float  aggBuf[8][DK];
    __shared__ float  battS[DK], w11S[DK], w22S[DK], w12S[DK];

    const int b    = blockIdx.x;
    const int tid  = threadIdx.x;
    const int wave = tid >> 5;
    const int lane = tid & 31;

    for (int idx = tid; idx < NTOT*DK; idx += 256) {
        int r = idx >> 6, c = idx & 63;
        Xs[r][c] = xh[(b*NTOT + r)*DK + c];
    }
    for (int idx = tid; idx < DK*DK; idx += 256) {
        int r = idx >> 6, c = idx & 63;
        Ws[r][c] = (__bf16)Watt[idx];
    }
    if (tid < DK) {
        battS[tid] = batt[tid];
        w11S[tid]  = w11[tid];
        w22S[tid]  = w22[tid];
        w12S[tid]  = w12[tid];
    }
    __syncthreads();

    const int i    = blockIdx.y * 8 + wave;        // this wave's target node
    const int oloc = lane & 15;
    const int khih = lane >> 4;                    // which 16-lane half

    // B fragments: M = Watt ⊙ x_i, transposed (k x o).  B k order used here (contiguous, same as A):
    // element e of lane L -> n = L%16, k = e + 16*(L/16)   (per 32-row K-slab)
    v16bf bfrag[4][2];
#pragma unroll
    for (int ot = 0; ot < 4; ++ot) {
        const int o = ot*16 + oloc;
#pragma unroll
        for (int kh = 0; kh < 2; ++kh) {
#pragma unroll
            for (int e = 0; e < 16; ++e) {
                const int k = kh*32 + 16*khih + e;
                bfrag[ot][kh][e] = (__bf16)((float)Ws[o][k] * (float)Xs[i][k]);
            }
        }
    }

    const v8f vz = {0.f,0.f,0.f,0.f,0.f,0.f,0.f,0.f};

    for (int jt = 0; jt < 16; ++jt) {
        // A fragments (16x32 bf16).  A k order used here (contiguous, same as B): lanes 0-15 -> M=lane,
        // K(lane,e) = e + 16*(lane>=16)
        const int m = jt*16 + oloc;
        v16bf a0, a1;
#pragma unroll
        for (int e = 0; e < 16; ++e) {
            const int k = e + 16*khih;
            a0[e] = Xs[m][k];
            a1[e] = Xs[m][32 + k];
        }
        v8f acc[4];
#pragma unroll
        for (int ot = 0; ot < 4; ++ot) {
            acc[ot] = __builtin_amdgcn_wmma_f32_16x16x32_bf16(
                false, a0, false, bfrag[ot][0], (short)0, vz, false, false);
            acc[ot] = __builtin_amdgcn_wmma_f32_16x16x32_bf16(
                false, a1, false, bfrag[ot][1], (short)0, acc[ot], false, false);
        }
        // quadrant weight: (i<128,j<128)->w11  (i>=128,j>=128)->w22  else w12
        const float* wq = ((i < 128) == (jt < 8)) ? ((i < 128) ? w11S : w22S) : w12S;
        float wv[4], bv[4];
#pragma unroll
        for (int ot = 0; ot < 4; ++ot) {
            const int o = ot*16 + oloc;
            wv[ot] = wq[o];
            bv[ot] = battS[o];
        }
#pragma unroll
        for (int r = 0; r < 8; ++r) {
            float p = 0.f;
#pragma unroll
            for (int ot = 0; ot < 4; ++ot)
                p += fast_tanh(acc[ot][r] + bv[ot]) * wv[ot];
            // reduce over the 16 o-lanes of this half (masks stay below bit 4)
            p += __shfl_xor(p, 1); p += __shfl_xor(p, 2);
            p += __shfl_xor(p, 4); p += __shfl_xor(p, 8);
            if (oloc == 0) sBuf[wave][jt*16 + r + 8*khih] = p;
        }
    }
    __syncthreads();

    // softmax over j (256 scores per wave)
    float sv[8];
#pragma unroll
    for (int t = 0; t < 8; ++t) sv[t] = sBuf[wave][lane + 32*t];
    float mx = sv[0];
#pragma unroll
    for (int t = 1; t < 8; ++t) mx = fmaxf(mx, sv[t]);
#pragma unroll
    for (int d = 1; d < 32; d <<= 1) mx = fmaxf(mx, __shfl_xor(mx, d));
    float sum = 0.f;
#pragma unroll
    for (int t = 0; t < 8; ++t) { sv[t] = __expf(sv[t] - mx); sum += sv[t]; }
#pragma unroll
    for (int d = 1; d < 32; d <<= 1) sum += __shfl_xor(sum, d);
    const float inv = 1.f / sum;
#pragma unroll
    for (int t = 0; t < 8; ++t) sBuf[wave][lane + 32*t] = sv[t] * inv;
    __syncthreads();

    // agg[i,:] = amap @ X_b : adjacent dims per lane so the two bf16 LDS
    // reads fuse into one ds_load_b32 per j.
    const int d0 = 2*lane, d1 = 2*lane + 1;
    float ag0 = 0.f, ag1 = 0.f;
#pragma unroll 4
    for (int j = 0; j < NTOT; ++j) {
        const float p = sBuf[wave][j];
        ag0 += p * (float)Xs[j][d0];
        ag1 += p * (float)Xs[j][d1];
    }
    aggBuf[wave][d0] = ag0;
    aggBuf[wave][d1] = ag1;
    __syncthreads();

    // h = agg@Wpa^T + bpa + x_i@Wpo^T + bpo   (float4 weight rows)
#pragma unroll
    for (int half = 0; half < 2; ++half) {
        const int o = lane + 32*half;
        const float4* Wa4 = (const float4*)(Wpa + o*DK);
        const float4* Wo4 = (const float4*)(Wpo + o*DK);
        float accv = bpa[o] + bpo[o];
#pragma unroll 4
        for (int q = 0; q < DK/4; ++q) {
            const float4 wa = Wa4[q];
            const float4 wo = Wo4[q];
            const int k = 4*q;
            accv += aggBuf[wave][k+0]*wa.x + aggBuf[wave][k+1]*wa.y
                  + aggBuf[wave][k+2]*wa.z + aggBuf[wave][k+3]*wa.w
                  + (float)Xs[i][k+0]*wo.x + (float)Xs[i][k+1]*wo.y
                  + (float)Xs[i][k+2]*wo.z + (float)Xs[i][k+3]*wo.w;
        }
        h_ws[(b*NTOT + i)*DK + o] = accv;
    }
}

// ---------------------------------------------------------------- master path
__global__ __launch_bounds__(256) void master_kernel(
        const float* __restrict__ x_f, const float* __restrict__ master,
        const float* __restrict__ WattM, const float* __restrict__ battM,
        const float* __restrict__ wM,
        const float* __restrict__ WpaM, const float* __restrict__ bpaM,
        const float* __restrict__ WpoM, const float* __restrict__ bpoM,
        float* __restrict__ out_master) {
    __shared__ float WmS[DK][DK+1];
    __shared__ float mS[DK], wMS[DK], battS[DK], mgS[DK];
    __shared__ float red[256], am[256];
    const int b = blockIdx.x, tid = threadIdx.x;

    for (int idx = tid; idx < DK*DK; idx += 256) WmS[idx >> 6][idx & 63] = WattM[idx];
    if (tid < DK) { mS[tid] = master[b*DK + tid]; wMS[tid] = wM[tid]; battS[tid] = battM[tid]; }
    __syncthreads();

    float xm[DK];
    const float* xr = x_f + (b*NTOT + tid)*DK;
#pragma unroll 8
    for (int k = 0; k < DK; ++k) xm[k] = xr[k] * mS[k];
    float s = 0.f;
    for (int o = 0; o < DK; ++o) {
        float acc = battS[o];
#pragma unroll 8
        for (int k = 0; k < DK; ++k) acc += xm[k] * WmS[o][k];
        s += fast_tanh(acc) * wMS[o];
    }
    red[tid] = s; __syncthreads();
    for (int off = 128; off > 0; off >>= 1) {
        if (tid < off) red[tid] = fmaxf(red[tid], red[tid + off]);
        __syncthreads();
    }
    const float mx = red[0]; __syncthreads();
    const float e = __expf(s - mx);
    red[tid] = e; __syncthreads();
    for (int off = 128; off > 0; off >>= 1) {
        if (tid < off) red[tid] += red[tid + off];
        __syncthreads();
    }
    const float inv = 1.f / red[0]; __syncthreads();
    am[tid] = e * inv; __syncthreads();

    if (tid < DK) {
        float mg = 0.f;
        for (int n = 0; n < NTOT; ++n) mg += am[n] * x_f[(b*NTOT + n)*DK + tid];
        mgS[tid] = mg;
    }
    __syncthreads();
    if (tid < DK) {
        const int o = tid;
        float acc = bpaM[o] + bpoM[o];
        for (int k = 0; k < DK; ++k)
            acc += mgS[k] * WpaM[o*DK + k] + mS[k] * WpoM[o*DK + k];
        out_master[b*DK + o] = acc;
    }
}

// ---------------------------------------------------------------- BN stats
__global__ __launch_bounds__(256) void bnstats_kernel(const float* __restrict__ h_ws,
                                                      float* __restrict__ mu,
                                                      float* __restrict__ rstd) {
    __shared__ float rs[256], rs2[256];
    const int c = blockIdx.x, tid = threadIdx.x;
    float s = 0.f, s2 = 0.f;
    for (int r = tid; r < ROWS; r += 256) {
        const float v = h_ws[r*DK + c];
        s += v; s2 += v*v;
    }
    rs[tid] = s; rs2[tid] = s2; __syncthreads();
    for (int off = 128; off > 0; off >>= 1) {
        if (tid < off) { rs[tid] += rs[tid + off]; rs2[tid] += rs2[tid + off]; }
        __syncthreads();
    }
    if (tid == 0) {
        const float m = rs[0] * (1.f/(float)ROWS);
        const float v = rs2[0] * (1.f/(float)ROWS) - m*m;
        mu[c]   = m;
        rstd[c] = rsqrtf(v + 1e-5f);
    }
}

// ---------------------------------------------------------------- BN + SiLU
__global__ void final_kernel(const float* __restrict__ h_ws,
                             const float* __restrict__ mu, const float* __restrict__ rstd,
                             const float* __restrict__ gamma, const float* __restrict__ beta,
                             float* __restrict__ out) {
    int idx = blockIdx.x * blockDim.x + threadIdx.x;
    if (idx >= HELEMS) return;
    int o = idx & 63, row = idx >> 6, n = row & 255, b = row >> 8;
    const float v = (h_ws[idx] - mu[o]) * rstd[o] * gamma[o] + beta[o];
    const float y = v / (1.f + __expf(-v));     // SiLU
    const int dst = (n < 128) ? ((b*128 + n)*DK + o)
                              : (BATCH*128*DK + (b*128 + (n-128))*DK + o);
    out[dst] = y;
}

// ---------------------------------------------------------------------------
extern "C" void kernel_launch(void* const* d_in, const int* in_sizes, int n_in,
                              void* d_out, int out_size, void* d_ws, size_t ws_size,
                              hipStream_t stream) {
    const float* x1    = (const float*)d_in[0];
    const float* x2    = (const float*)d_in[1];
    const float* Wt1   = (const float*)d_in[2];
    const float* bt1   = (const float*)d_in[3];
    const float* Wt2   = (const float*)d_in[4];
    const float* bt2   = (const float*)d_in[5];
    const float* Watt  = (const float*)d_in[6];
    const float* batt  = (const float*)d_in[7];
    const float* WattM = (const float*)d_in[8];
    const float* battM = (const float*)d_in[9];
    const float* w11   = (const float*)d_in[10];
    const float* w22   = (const float*)d_in[11];
    const float* w12   = (const float*)d_in[12];
    const float* wM    = (const float*)d_in[13];
    const float* Wpa   = (const float*)d_in[14];
    const float* bpa   = (const float*)d_in[15];
    const float* Wpo   = (const float*)d_in[16];
    const float* bpo   = (const float*)d_in[17];
    const float* WpaM  = (const float*)d_in[18];
    const float* bpaM  = (const float*)d_in[19];
    const float* WpoM  = (const float*)d_in[20];
    const float* bpoM  = (const float*)d_in[21];
    const float* gamma = (const float*)d_in[22];
    const float* beta  = (const float*)d_in[23];

    char* ws = (char*)d_ws;
    float*  x_f    = (float*)(ws);                     // 1,048,576 B
    __bf16* xh     = (__bf16*)(ws + 1048576);          //   524,288 B
    float*  master = (float*)(ws + 1572864);           //     4,096 B
    float*  h_ws   = (float*)(ws + 1576960);           // 1,048,576 B
    float*  mu     = (float*)(ws + 2625536);           //       256 B
    float*  rstd   = (float*)(ws + 2625792);           //       256 B
    float*  outF   = (float*)d_out;

    proj_kernel<<<HELEMS/256, 256, 0, stream>>>(x1, x2, Wt1, bt1, Wt2, bt2, x_f, xh);
    mean_kernel<<<BATCH, 64, 0, stream>>>(x_f, master);
    attn_kernel<<<dim3(BATCH, NTOT/8), 256, 0, stream>>>(
        xh, Watt, batt, w11, w22, w12, Wpa, bpa, Wpo, bpo, h_ws);
    master_kernel<<<BATCH, 256, 0, stream>>>(
        x_f, master, WattM, battM, wM, WpaM, bpaM, WpoM, bpoM,
        outF + 2*BATCH*128*DK);
    bnstats_kernel<<<DK, 256, 0, stream>>>(h_ws, mu, rstd);
    final_kernel<<<HELEMS/256, 256, 0, stream>>>(h_ws, mu, rstd, gamma, beta, outF);
}
